// RNN_46789373723334
// MI455X (gfx1250) — matrix operand order repaired and checked by reading, not compiled, not run on hardware
//
#include <hip/hip_runtime.h>
#include <math.h>

typedef _Float16 v8h  __attribute__((ext_vector_type(8)));
typedef _Float16 v16h __attribute__((ext_vector_type(16)));
typedef float    v8f  __attribute__((ext_vector_type(8)));

#define WMMA_F16(a, b, c) \
    __builtin_amdgcn_wmma_f32_16x16x32_f16(false, (a), false, (b), (short)0, (c), false, false)

// Hardware tanh if the toolchain exposes it (gfx1250 has a TANH trans op),
// otherwise ocml tanhf.
__device__ __forceinline__ float fast_tanh(float x) {
#if __has_builtin(__builtin_amdgcn_tanhf)
    return __builtin_amdgcn_tanhf(x);
#elif __has_builtin(__builtin_amdgcn_tanh_f32)
    return __builtin_amdgcn_tanh_f32(x);
#else
    return tanhf(x);
#endif
}

// A-fragment (16x32 f16, MxK) in this file's contiguous k order (B frags match):
// lane = (sel<<4)|ml holds row M=ml, K = sel*16..sel*16+15 (contiguous)
__device__ __forceinline__ v16h make_a_frag(const _Float16* rp, int k0, int sel) {
    const v8h lo = *(const v8h*)(rp + k0 + sel * 16);
    const v8h hi = *(const v8h*)(rp + k0 + sel * 16 + 8);
    return __builtin_shufflevector(lo, hi, 0, 1, 2, 3, 4, 5, 6, 7,
                                   8, 9, 10, 11, 12, 13, 14, 15);
}

// ---------------- f32 -> f16 convert ----------------
__global__ void cvt_f32_f16(const float* __restrict__ s, _Float16* __restrict__ d, int n) {
    int i = blockIdx.x * blockDim.x + threadIdx.x;
    if (i < n) d[i] = (_Float16)s[i];
}

// ---------------- generic tiled WMMA GEMM ----------------
// Rows m = t*32 + b (t = m>>5, b = m&31), A row at  A + t*stride_t + b*stride_b.
// W is row-major [n_total x K] (f16); computes C = A * W^T.
// MODE 0: out f16 at out[m*n_total + h], bias = bias1[h] + bias2[h]   (pre-activations)
// MODE 1: out f32 at out[b*128*512 + h*512 + t], bias = bias1[h]       (FC head, [B,O,T])
template <int K, int MODE>
__global__ void __launch_bounds__(128)
gemm16(const _Float16* __restrict__ A, const _Float16* __restrict__ W,
       const float* __restrict__ bias1, const float* __restrict__ bias2,
       void* __restrict__ outp, int stride_t, int stride_b,
       int tiles_n, int n_total) {
    const int wave   = blockIdx.x * (blockDim.x >> 5) + (threadIdx.x >> 5);
    const int lane   = threadIdx.x & 31;
    const int ml     = lane & 15;
    const int sel    = lane >> 4;
    const int tile_m = wave / tiles_n;
    const int tile_n = wave % tiles_n;

    const int m = tile_m * 16 + ml;
    const _Float16* rp = A + (m >> 5) * stride_t + (m & 31) * stride_b;
    const _Float16* wp = W + (tile_n * 16 + ml) * K;

    v8f acc = {0.f, 0.f, 0.f, 0.f, 0.f, 0.f, 0.f, 0.f};
#pragma unroll
    for (int k0 = 0; k0 < K; k0 += 32) {
        v16h af = make_a_frag(rp, k0, sel);
        v16h bf = *(const v16h*)(wp + k0 + sel * 16);  // B frag: col=ml, K=sel*16..+15
        acc = WMMA_F16(af, bf, acc);
    }

    const int h = tile_n * 16 + ml;
    if (MODE == 0) {
        const float bb = bias1[h] + bias2[h];
        _Float16* out = (_Float16*)outp;
#pragma unroll
        for (int r = 0; r < 8; ++r) {
            const int mo = tile_m * 16 + r + 8 * sel;  // C: vgpr r -> M = r + 8*sel
            out[mo * n_total + h] = (_Float16)(acc[r] + bb);
        }
    } else {
        const float bb = bias1[h];
        float* out = (float*)outp;
#pragma unroll
        for (int r = 0; r < 8; ++r) {
            const int mo = tile_m * 16 + r + 8 * sel;
            const int t = mo >> 5, b = mo & 31;
            out[b * (128 * 512) + h * 512 + t] = acc[r] + bb;  // [B,O,T]
        }
    }
}

// ---------------- sequential recurrence: h_t = tanh(pre_t + h_{t-1} W_hh^T) ----------------
// One workgroup = 16 waves on one WGP; wave w owns output columns h = w*32 .. w*32+31
// (two 16-wide n-tiles).  W_hh split:
//   k = 0..255   -> LDS (512 rows x 264-half padded rows, 270,336 B, loaded once)
//   k = 256..511 -> registers (2 n-tiles x 8 k-steps x v16h = 128 VGPRs per wave)
// h state: single LDS buffer (32 x 520 halves), two barriers per step.
__global__ void __launch_bounds__(512) rnn_rec(const _Float16* __restrict__ pre,
                                               const _Float16* __restrict__ Whh,
                                               _Float16* __restrict__ hout, int T) {
    __shared__ __align__(16) _Float16 WL[512 * 264];   // 270,336 B
    __shared__ __align__(16) _Float16 hbuf[32][520];   //  33,280 B  (total 303,616 B)

    const int tid  = threadIdx.x;
    const int w    = tid >> 5;   // 0..15
    const int lane = tid & 31;
    const int ml   = lane & 15;
    const int sel  = lane >> 4;

    // cooperative load of W_hh[:, 0:256] into LDS (16-byte chunks, padded rows)
    for (int c = tid; c < 512 * 32; c += 512) {
        const int row = c >> 5, c8 = c & 31;
        *(v8h*)(WL + row * 264 + c8 * 8) = *(const v8h*)(Whh + row * 512 + c8 * 8);
    }
    // zero the h state
    for (int i = tid; i < 32 * 520; i += 512) ((_Float16*)hbuf)[i] = (_Float16)0.f;

    // register-resident W_hh[:, 256:512] for this wave's two n-tiles
    v16h wreg[2][8];
#pragma unroll
    for (int nt = 0; nt < 2; ++nt) {
#pragma unroll
        for (int ks = 0; ks < 8; ++ks) {
            wreg[nt][ks] = *(const v16h*)(Whh + (w * 32 + nt * 16 + ml) * 512 +
                                          256 + ks * 32 + sel * 16);
        }
    }
    __syncthreads();

    for (int t = 0; t < T; ++t) {
        // Prefetch this step's 32 KB pre-activation slab (256 x 128B lines) so the
        // epilogue's scattered loads hit near caches.  Wave-uniform branch: EXEC is
        // all-ones again before any WMMA below.
        if (tid < 256) {
            __builtin_prefetch(pre + t * (32 * 512) + tid * 64);
        }

        v8f c[2][2];
#pragma unroll
        for (int nt = 0; nt < 2; ++nt)
#pragma unroll
            for (int mt = 0; mt < 2; ++mt)
                c[nt][mt] = (v8f){0.f, 0.f, 0.f, 0.f, 0.f, 0.f, 0.f, 0.f};

#pragma unroll
        for (int ks = 0; ks < 16; ++ks) {
            const int k0 = ks * 32;
            v16h a0 = make_a_frag(&hbuf[ml][0], k0, sel);        // batches 0..15
            v16h a1 = make_a_frag(&hbuf[16 + ml][0], k0, sel);   // batches 16..31
#pragma unroll
            for (int nt = 0; nt < 2; ++nt) {
                v16h bf;
                if (ks < 8) {
                    // lower K-half from LDS: col = ml, K = k0 + sel*16 .. +15
                    bf = *(const v16h*)(WL + (w * 32 + nt * 16 + ml) * 264 +
                                        k0 + sel * 16);
                } else {
                    bf = wreg[nt][ks - 8];                        // upper K-half: registers
                }
                c[nt][0] = WMMA_F16(a0, bf, c[nt][0]);
                c[nt][1] = WMMA_F16(a1, bf, c[nt][1]);
            }
        }

        __syncthreads();  // all reads of hbuf complete before overwriting

        const _Float16* pt = pre + t * (32 * 512);
        _Float16* ho = hout + t * (32 * 512);
#pragma unroll
        for (int nt = 0; nt < 2; ++nt) {
            const int h = w * 32 + nt * 16 + ml;
#pragma unroll
            for (int r = 0; r < 8; ++r) {
                const int b0 = r + 8 * sel;   // C: vgpr r -> M = r + 8*sel
                _Float16 v0 = (_Float16)fast_tanh(c[nt][0][r] + (float)pt[b0 * 512 + h]);
                hbuf[b0][h] = v0;
                ho[b0 * 512 + h] = v0;
                const int b1 = b0 + 16;
                _Float16 v1 = (_Float16)fast_tanh(c[nt][1][r] + (float)pt[b1 * 512 + h]);
                hbuf[b1][h] = v1;
                ho[b1 * 512 + h] = v1;
            }
        }
        __syncthreads();  // new h visible to all waves
    }
}

// ---------------- host ----------------
extern "C" void kernel_launch(void* const* d_in, const int* in_sizes, int n_in,
                              void* d_out, int out_size, void* d_ws, size_t ws_size,
                              hipStream_t stream) {
    (void)in_sizes; (void)n_in; (void)out_size; (void)ws_size;

    const float* X     = (const float*)d_in[0];   // [32,512,128]
    const float* W_ih0 = (const float*)d_in[1];   // [512,128]
    const float* W_hh0 = (const float*)d_in[2];   // [512,512]
    const float* b_ih0 = (const float*)d_in[3];
    const float* b_hh0 = (const float*)d_in[4];
    const float* W_ih1 = (const float*)d_in[5];   // [512,512]
    const float* W_hh1 = (const float*)d_in[6];   // [512,512]
    const float* b_ih1 = (const float*)d_in[7];
    const float* b_hh1 = (const float*)d_in[8];
    const float* W_fc  = (const float*)d_in[9];   // [128,512]
    const float* b_fc  = (const float*)d_in[10];

    const int B = 32, T = 512, I = 128, H = 512;

    char* p = (char*)d_ws;
    _Float16* X16   = (_Float16*)p; p += (size_t)B * T * I * 2;   // 4 MB
    _Float16* Wih0h = (_Float16*)p; p += (size_t)H * I * 2;       // 128 KB
    _Float16* Whh0h = (_Float16*)p; p += (size_t)H * H * 2;       // 512 KB
    _Float16* Wih1h = (_Float16*)p; p += (size_t)H * H * 2;       // 512 KB
    _Float16* Whh1h = (_Float16*)p; p += (size_t)H * H * 2;       // 512 KB
    _Float16* Wfch  = (_Float16*)p; p += (size_t)128 * H * 2;     // 128 KB
    _Float16* pre16 = (_Float16*)p; p += (size_t)T * B * H * 2;   // 16 MB (reused L0/L1)
    _Float16* h1    = (_Float16*)p; p += (size_t)T * B * H * 2;   // 16 MB
    _Float16* h2    = (_Float16*)p; p += (size_t)T * B * H * 2;   // 16 MB

    auto cvt = [&](const float* s, _Float16* d, int n) {
        cvt_f32_f16<<<(n + 255) / 256, 256, 0, stream>>>(s, d, n);
    };
    cvt(X, X16, B * T * I);
    cvt(W_ih0, Wih0h, H * I);
    cvt(W_hh0, Whh0h, H * H);
    cvt(W_ih1, Wih1h, H * H);
    cvt(W_hh1, Whh1h, H * H);
    cvt(W_fc, Wfch, 128 * H);

    // pre0 = X·W_ih0^T + b_ih0 + b_hh0 ; rows m=t*32+b, X row at b*T*I + t*I
    gemm16<128, 0><<<8192, 128, 0, stream>>>(X16, Wih0h, b_ih0, b_hh0, pre16,
                                             /*stride_t=*/I, /*stride_b=*/T * I,
                                             /*tiles_n=*/32, /*n_total=*/H);
    // layer-0 recurrence -> h1 [T,B,H]
    rnn_rec<<<1, 512, 0, stream>>>(pre16, Whh0h, h1, T);

    // pre1 = h1·W_ih1^T + b_ih1 + b_hh1 ; h1 row at t*B*H + b*H
    gemm16<512, 0><<<8192, 128, 0, stream>>>(h1, Wih1h, b_ih1, b_hh1, pre16,
                                             /*stride_t=*/B * H, /*stride_b=*/H,
                                             /*tiles_n=*/32, /*n_total=*/H);
    // layer-1 recurrence -> h2
    rnn_rec<<<1, 512, 0, stream>>>(pre16, Whh1h, h2, T);

    // FC head: out[b,o,t] = h2·W_fc^T + b_fc
    gemm16<512, 1><<<2048, 128, 0, stream>>>(h2, Wfch, b_fc, b_fc, d_out,
                                             /*stride_t=*/B * H, /*stride_b=*/H,
                                             /*tiles_n=*/8, /*n_total=*/128);
}
